// HDCBrainV14_1_37288906064328
// MI455X (gfx1250) — matrix operand order repaired and checked by reading, not compiled, not run on hardware
//
#include <hip/hip_runtime.h>
#include <hip/hip_bf16.h>

#define BB 4
#define TT 512
#define DD 2048
#define HH 4
#define HD 512
#define LL 8
#define INNER 2560
#define VV 32000
#define BT (BB * TT)

typedef __attribute__((ext_vector_type(16))) _Float16 v16h;
typedef __attribute__((ext_vector_type(8)))  _Float16 v8h;
typedef __attribute__((ext_vector_type(4)))  _Float16 v4h;
typedef __attribute__((ext_vector_type(8)))  float    v8f;
typedef __attribute__((ext_vector_type(4)))  float    v4f;

__device__ __forceinline__ v8f wmma_f16(v16h a, v16h b, v8f c) {
  // D = A(16x32 f16) x B(32x16 f16) + C(16x16 f32)
  return __builtin_amdgcn_wmma_f32_16x16x32_f16(
      /*neg_a=*/false, a, /*neg_b=*/false, b,
      /*c_mod=*/(short)0, c, /*reuse_a=*/false, /*reuse_b=*/false);
}

__device__ __forceinline__ float block_reduce_sum(float v, float* sbuf) {
  int tid = threadIdx.x;
  sbuf[tid] = v;
  __syncthreads();
  for (int off = 128; off > 0; off >>= 1) {
    if (tid < off) sbuf[tid] += sbuf[tid + off];
    __syncthreads();
  }
  float r = sbuf[0];
  __syncthreads();
  return r;
}

// ---------------------------------------------------------------------------
// Embedding: STE bipolar encode (forward = alpha*sign) + cyclic permutation
// ---------------------------------------------------------------------------
__global__ void __launch_bounds__(256)
embed_kernel(const int* __restrict__ idx, const float* __restrict__ codebook,
             float* __restrict__ h) {
  __shared__ float sbuf[256];
  int bt = blockIdx.x;
  int t = bt % TT;
  int tok = idx[bt];
  const float* row = codebook + (size_t)tok * DD;
  int tid = threadIdx.x;
  float s = 0.f;
  for (int d = tid; d < DD; d += 256) s += fabsf(row[d]);
  float alpha = block_reduce_sum(s, sbuf) * (1.0f / DD);
  for (int d = tid; d < DD; d += 256) {
    int src = d - t; if (src < 0) src += DD;   // (d - t) mod D
    float rv = row[src];
    h[(size_t)bt * DD + d] = (rv < 0.f) ? -alpha : alpha;  // sign(0) -> +alpha
  }
}

// ---------------------------------------------------------------------------
// HDCMemory gates: mass = sigmoid(h.mass_w), logdecay = log(clip(sigmoid(h.decay_w)))
// ---------------------------------------------------------------------------
__global__ void __launch_bounds__(256)
gates_kernel(const float* __restrict__ h, const float* __restrict__ mw,
             const float* __restrict__ dw, float* __restrict__ weighted,
             float* __restrict__ logdecay) {
  __shared__ float sbuf[256];
  int bt = blockIdx.x, tid = threadIdx.x;
  const v4f* hr = (const v4f*)(h + (size_t)bt * DD);
  const v4f* m4 = (const v4f*)mw;
  const v4f* d4 = (const v4f*)dw;
  float sm = 0.f, sd = 0.f;
#pragma unroll
  for (int j = 0; j < DD / 1024; ++j) {   // DD/4 vectors, 256 threads
    int d = tid + j * 256;
    v4f x = hr[d], mm = m4[d], dd4 = d4[d];
    sm += x[0]*mm[0] + x[1]*mm[1] + x[2]*mm[2] + x[3]*mm[3];
    sd += x[0]*dd4[0] + x[1]*dd4[1] + x[2]*dd4[2] + x[3]*dd4[3];
  }
  float m  = block_reduce_sum(sm, sbuf);
  float dd = block_reduce_sum(sd, sbuf);
  float mass = 1.f / (1.f + __expf(-m));
  float dec  = 1.f / (1.f + __expf(-dd));
  dec = fmaxf(dec, 1e-6f);
  if (tid == 0) logdecay[bt] = __logf(dec);
  v4f* w4 = (v4f*)(weighted + (size_t)bt * DD);
#pragma unroll
  for (int j = 0; j < DD / 1024; ++j) {
    int d = tid + j * 256;
    v4f x = hr[d];
    w4[d] = x * mass;
  }
}

__global__ void cumsum_kernel(const float* __restrict__ ld, float* __restrict__ cum) {
  if (threadIdx.x != 0) return;
  int b = blockIdx.x;
  float s = 0.f;
  for (int t = 0; t < TT; ++t) { s += ld[b * TT + t]; cum[b * TT + t] = s; }
}

// ---------------------------------------------------------------------------
// mem[b,t,:] = sum_{s<=t} exp(cum[t]-cum[s]) * weighted[b,s,:]   (WMMA over s)
// grid (D/128, T/16, B), 8 waves/block, each wave a 16x16 (t x d) tile.
// cum[] row staged into LDS via CDNA5 async DMA (ASYNCcnt-tracked).
// ---------------------------------------------------------------------------
__global__ void __launch_bounds__(256)
mem_kernel(const float* __restrict__ weighted, const float* __restrict__ cum,
           float* __restrict__ memout) {
  __shared__ float scum[TT];
  int b = blockIdx.z;
  int t0 = blockIdx.y * 16;
  int wave = threadIdx.x >> 5, lane = threadIdx.x & 31;
  int d0 = blockIdx.x * 128 + wave * 16;

  // Async global->LDS DMA: 128 lanes x b128 = the whole 2KB cum row.
  // LDS address operand = low 32 bits of the generic LDS pointer (lds offset).
  if (threadIdx.x < TT / 4) {
    unsigned lds_off = (unsigned)(size_t)(&scum[threadIdx.x * 4]);
    const float* gp = cum + (size_t)b * TT + threadIdx.x * 4;
    asm volatile("global_load_async_to_lds_b128 %0, %1, off"
                 :: "v"(lds_off), "v"(gp)
                 : "memory");
  }
  asm volatile("s_wait_asynccnt 0x0" ::: "memory");
  __syncthreads();

  int grp = lane >> 4, lane16 = lane & 15;
  int trow = t0 + lane16;
  float ct = scum[trow];
  int dcol = d0 + lane16;
  const float* wbase = weighted + ((size_t)b * TT) * DD + dcol;
  v8f acc = {};
  int slimit = t0 + 16;
  for (int s0 = 0; s0 < slimit; s0 += 32) {
    int sa = s0 + 8 * grp;
    const float* pb = wbase + (size_t)(s0 + 8 * grp) * DD;
    v16h a, bb;
#pragma unroll
    for (int i = 0; i < 8; ++i) {
      float e0 = __expf(ct - scum[sa + i]);        // diff<=0: cum is decreasing
      float e1 = __expf(ct - scum[sa + i + 16]);
      a[i]     = (_Float16)((sa + i      <= trow) ? e0 : 0.f);
      a[i + 8] = (_Float16)((sa + i + 16 <= trow) ? e1 : 0.f);
    }
#pragma unroll
    for (int i = 0; i < 16; ++i) bb[i] = (_Float16)pb[(size_t)(i + (i & 8)) * DD];
    acc = wmma_f16(a, bb, acc);
  }
  float* obase = memout + ((size_t)(b * TT + t0)) * DD + dcol;
#pragma unroll
  for (int j = 0; j < 8; ++j) obase[(size_t)(j + 8 * grp) * DD] = acc[j];
}

// ---------------------------------------------------------------------------
// out = LN(x (+res)) * w + b (+post).  One 256-thread block per row, v4f traffic.
// ---------------------------------------------------------------------------
__global__ void __launch_bounds__(256)
add_ln_kernel(const float* __restrict__ x, const float* __restrict__ res,
              const float* __restrict__ w, const float* __restrict__ bvec,
              const float* __restrict__ post, float* __restrict__ out) {
  __shared__ float sbuf[256];
  int bt = blockIdx.x, tid = threadIdx.x;
  const v4f* xr = (const v4f*)(x + (size_t)bt * DD);
  const v4f* rr = res ? (const v4f*)(res + (size_t)bt * DD) : nullptr;
  v4f loc[DD / 1024];
  float s = 0.f;
#pragma unroll
  for (int j = 0; j < DD / 1024; ++j) {
    int d = tid + j * 256;
    v4f v = xr[d];
    if (rr) v += rr[d];
    loc[j] = v;
    s += v[0] + v[1] + v[2] + v[3];
  }
  float mean = block_reduce_sum(s, sbuf) * (1.f / DD);
  float s2 = 0.f;
#pragma unroll
  for (int j = 0; j < DD / 1024; ++j) {
    v4f dv = loc[j] - mean;
    s2 += dv[0]*dv[0] + dv[1]*dv[1] + dv[2]*dv[2] + dv[3]*dv[3];
  }
  float var = block_reduce_sum(s2, sbuf) * (1.f / DD);
  float inv = rsqrtf(var + 1e-5f);
  const v4f* w4 = (const v4f*)w;
  const v4f* b4 = (const v4f*)bvec;
  const v4f* p4 = post ? (const v4f*)post : nullptr;
  v4f* o4 = (v4f*)(out + (size_t)bt * DD);
#pragma unroll
  for (int j = 0; j < DD / 1024; ++j) {
    int d = tid + j * 256;
    v4f o = (loc[j] - mean) * inv * w4[d] + b4[d];
    if (p4) o += p4[d];
    o4[d] = o;
  }
}

// ---------------------------------------------------------------------------
// STE binarize per-head q/k/v -> f16 [3,H,HD]
// ---------------------------------------------------------------------------
__global__ void __launch_bounds__(256)
ste_kernel(const float* __restrict__ q, const float* __restrict__ k,
           const float* __restrict__ v, _Float16* __restrict__ out) {
  __shared__ float sbuf[256];
  int which = blockIdx.x / HH, hh = blockIdx.x % HH, tid = threadIdx.x;
  const float* src = (which == 0 ? q : which == 1 ? k : v) + (size_t)hh * HD;
  float s = 0.f;
  for (int d = tid; d < HD; d += 256) s += fabsf(src[d]);
  float alpha = block_reduce_sum(s, sbuf) * (1.f / HD);
  for (int d = tid; d < HD; d += 256)
    out[((size_t)which * HH + hh) * HD + d] = (_Float16)((src[d] < 0.f) ? -alpha : alpha);
}

// ---------------------------------------------------------------------------
// attn[b,h,t,s] = causal ? sigmoid(4*scale*sum_hd (x_t*q)(x_s*k)) : 0  (WMMA over hd)
// grid (T/16, T/16/8, B*H); each wave one 16x16 (t x s) tile
// ---------------------------------------------------------------------------
__global__ void __launch_bounds__(256)
attn_score_kernel(const float* __restrict__ h, const _Float16* __restrict__ qkv,
                  float* __restrict__ attn) {
  int bh = blockIdx.z; int b = bh / HH; int head = bh % HH;
  int t0 = blockIdx.x * 16;
  int wave = threadIdx.x >> 5, lane = threadIdx.x & 31;
  int s0 = (blockIdx.y * 8 + wave) * 16;
  if (s0 > t0) return;                      // above causal diagonal at tile level
  int grp = lane >> 4, lane16 = lane & 15;
  const _Float16* qf = qkv + (size_t)head * HD;
  const _Float16* kf = qkv + ((size_t)HH + head) * HD;
  int trow = t0 + lane16, scol = s0 + lane16;
  const float* arow = h + ((size_t)(b * TT + trow)) * DD + head * HD;
  const float* brow = h + ((size_t)(b * TT + scol)) * DD + head * HD;
  v8f acc = {};
  for (int k0 = 0; k0 < HD; k0 += 32) {
    int ka = k0 + 8 * grp;
    v4f a0 = *(const v4f*)(arow + ka);
    v4f a1 = *(const v4f*)(arow + ka + 4);
    v4f a2 = *(const v4f*)(arow + ka + 16);
    v4f a3 = *(const v4f*)(arow + ka + 20);
    v8h q0 = *(const v8h*)(qf + ka);
    v8h q1 = *(const v8h*)(qf + ka + 16);
    v16h a;
#pragma unroll
    for (int i = 0; i < 4; ++i) {
      a[i]      = (_Float16)(a0[i] * (float)q0[i]);
      a[i + 4]  = (_Float16)(a1[i] * (float)q0[i + 4]);
      a[i + 8]  = (_Float16)(a2[i] * (float)q1[i]);
      a[i + 12] = (_Float16)(a3[i] * (float)q1[i + 4]);
    }
    int kb = k0 + 8 * grp;
    v4f b0 = *(const v4f*)(brow + kb);
    v4f b1 = *(const v4f*)(brow + kb + 4);
    v4f b2 = *(const v4f*)(brow + kb + 16);
    v4f b3 = *(const v4f*)(brow + kb + 20);
    v8h k0h = *(const v8h*)(kf + kb);
    v8h k1h = *(const v8h*)(kf + kb + 16);
    v16h bbf;
#pragma unroll
    for (int i = 0; i < 4; ++i) {
      bbf[i]      = (_Float16)(b0[i] * (float)k0h[i]);
      bbf[i + 4]  = (_Float16)(b1[i] * (float)k0h[i + 4]);
      bbf[i + 8]  = (_Float16)(b2[i] * (float)k1h[i]);
      bbf[i + 12] = (_Float16)(b3[i] * (float)k1h[i + 4]);
    }
    acc = wmma_f16(a, bbf, acc);
  }
  const float scale = 0.044194173824159216f;   // HD^-0.5, HD=512
  float* obase = attn + ((size_t)bh * TT + t0) * TT + scol;
#pragma unroll
  for (int j = 0; j < 8; ++j) {
    int tr = t0 + j + 8 * grp;
    float sv = acc[j] * scale;
    float av = (scol <= tr) ? 1.f / (1.f + __expf(-4.f * sv)) : 0.f;
    obase[(size_t)(j + 8 * grp) * TT] = av;
  }
}

// ---------------------------------------------------------------------------
// a_out[b,t,d] = sum_{s<=t} attn[b,h,t,s] * h[b,s,d] * v[d]   (WMMA over s)
// grid (D/128, T/16, B); loads unconditional (indices provably < T), value-masked.
// ---------------------------------------------------------------------------
__global__ void __launch_bounds__(256)
attn_out_kernel(const float* __restrict__ h, const float* __restrict__ attn,
                const _Float16* __restrict__ qkv, float* __restrict__ aout) {
  int b = blockIdx.z;
  int t0 = blockIdx.y * 16;
  int wave = threadIdx.x >> 5, lane = threadIdx.x & 31;
  int d0 = blockIdx.x * 128 + wave * 16;
  int head = d0 >> 9;                        // d0 / HD
  int grp = lane >> 4, lane16 = lane & 15;
  int trow = t0 + lane16, dcol = d0 + lane16;
  const _Float16* vf = qkv + (size_t)(2 * HH) * HD;   // [H,HD] flattened = D
  float vcol = (float)vf[dcol];
  const float* arow = attn + ((size_t)(b * HH + head) * TT + trow) * TT;
  const float* bcol = h + ((size_t)b * TT) * DD + dcol;
  v8f acc = {};
  int slimit = t0 + 16;
  for (int s0 = 0; s0 < slimit; s0 += 32) {
    int ka = s0 + 8 * grp;
    v4f r0 = *(const v4f*)(arow + ka);
    v4f r1 = *(const v4f*)(arow + ka + 4);
    v4f r2 = *(const v4f*)(arow + ka + 16);
    v4f r3 = *(const v4f*)(arow + ka + 20);
    v16h a;
#pragma unroll
    for (int i = 0; i < 4; ++i) {
      a[i]      = (_Float16)((ka + i      < slimit) ? r0[i] : 0.f);
      a[i + 4]  = (_Float16)((ka + i + 4  < slimit) ? r1[i] : 0.f);
      a[i + 8]  = (_Float16)((ka + i + 16 < slimit) ? r2[i] : 0.f);
      a[i + 12] = (_Float16)((ka + i + 20 < slimit) ? r3[i] : 0.f);
    }
    const float* pb = bcol + (size_t)(s0 + 8 * grp) * DD;
    v16h bbf;
#pragma unroll
    for (int i = 0; i < 16; ++i)
      bbf[i] = (_Float16)(pb[(size_t)(i + (i & 8)) * DD] * vcol);
    acc = wmma_f16(a, bbf, acc);
  }
  float* obase = aout + ((size_t)(b * TT + t0)) * DD + dcol;
#pragma unroll
  for (int j = 0; j < 8; ++j) obase[(size_t)(j + 8 * grp) * DD] = acc[j];
}

// ---------------------------------------------------------------------------
// Weight converters (f32 -> f16); NT loads for read-once codebook stream
// ---------------------------------------------------------------------------
__global__ void __launch_bounds__(256)
conv_f16_kernel(const float* __restrict__ src, _Float16* __restrict__ dst, size_t n4) {
  size_t i = (size_t)blockIdx.x * 256 + threadIdx.x;
  if (i >= n4) return;
  v4f f = __builtin_nontemporal_load((const v4f*)(src + i * 4));
  v4h o;
#pragma unroll
  for (int j = 0; j < 4; ++j) o[j] = (_Float16)f[j];
  *(v4h*)(dst + i * 4) = o;
}

__global__ void __launch_bounds__(256)
convt_f16_kernel(const float* __restrict__ src, _Float16* __restrict__ dst,
                 int K, int N) {  // src [K,N] -> dst [N,K], 4 k-elems per thread
  size_t i = ((size_t)blockIdx.x * 256 + threadIdx.x) * 4;  // dst flat index
  if (i >= (size_t)K * N) return;
  int n = (int)(i / K), k = (int)(i % K);
  const float* s = src + (size_t)k * N + n;
  v4h o;
#pragma unroll
  for (int j = 0; j < 4; ++j) o[j] = (_Float16)s[(size_t)j * N];
  *(v4h*)(dst + i) = o;
}

// ---------------------------------------------------------------------------
// C[M,N] = epilogue(A[M,K] f32 x Bt[N,K] f16)
// epilogue 0: C = acc*scale (NT store)  1: C = gelu(acc+bias)  2: C += acc+bias
// 8 waves/block, each wave 16x64 (4 WMMAs per A fragment), block tile 32x256
// ---------------------------------------------------------------------------
__global__ void __launch_bounds__(256)
gemm_kernel(const float* __restrict__ A, const _Float16* __restrict__ Bt,
            const float* __restrict__ bias, float* __restrict__ C,
            int M, int N, int K, int epilogue, const float* __restrict__ scale_ptr) {
  int wave = threadIdx.x >> 5, lane = threadIdx.x & 31;
  int grp = lane >> 4, lane16 = lane & 15;
  int m0 = blockIdx.y * 32 + (wave & 1) * 16;
  int n0 = blockIdx.x * 256 + (wave >> 1) * 64;
  int mrow = m0 + lane16;
  const float* arow = A + (size_t)mrow * K;
  const _Float16* bptr[4];
#pragma unroll
  for (int j = 0; j < 4; ++j)
    bptr[j] = Bt + (size_t)(n0 + j * 16 + lane16) * K + 16 * grp;
  v8f acc[4] = {};
  for (int k0 = 0; k0 < K; k0 += 32) {
    const float* p0 = arow + k0 + 16 * grp;
    v4f a0 = *(const v4f*)p0;
    v4f a1 = *(const v4f*)(p0 + 4);
    v4f a2 = *(const v4f*)(p0 + 8);
    v4f a3 = *(const v4f*)(p0 + 12);
    v16h a;
#pragma unroll
    for (int i = 0; i < 4; ++i) {
      a[i]      = (_Float16)a0[i];
      a[i + 4]  = (_Float16)a1[i];
      a[i + 8]  = (_Float16)a2[i];
      a[i + 12] = (_Float16)a3[i];
    }
#pragma unroll
    for (int j = 0; j < 4; ++j) {
      v16h bbf = *(const v16h*)(bptr[j] + k0);   // 32B-aligned f16 vector load
      acc[j] = wmma_f16(a, bbf, acc[j]);
    }
    if (k0 + 64 < K) __builtin_prefetch(arow + k0 + 64, 0, 1);  // global_prefetch_b8
  }
  float scl = scale_ptr ? *scale_ptr : 1.f;
#pragma unroll
  for (int j = 0; j < 4; ++j) {
    int ncol = n0 + j * 16 + lane16;
    float bv = bias ? bias[ncol] : 0.f;
#pragma unroll
    for (int r = 0; r < 8; ++r) {
      size_t oi = (size_t)(m0 + r + 8 * grp) * N + ncol;
      float v = acc[j][r];
      if (epilogue == 0) {
        __builtin_nontemporal_store(v * scl, &C[oi]);   // streamed logits
      } else if (epilogue == 1) {
        float xx = v + bv;
        C[oi] = 0.5f * xx * (1.f + erff(xx * 0.70710678118654752f));
      } else {
        C[oi] += v + bv;
      }
    }
  }
}

__global__ void __launch_bounds__(256)
combine_kernel(float* __restrict__ hbuf, const float* __restrict__ ht,
               const float* __restrict__ gates) {
  float g = 1.f / (1.f + __expf(-gates[1]));
  size_t i = (size_t)blockIdx.x * 256 + threadIdx.x;
  if (i < (size_t)BT * DD) { float hv = hbuf[i]; hbuf[i] = hv + g * (ht[i] - hv); }
}

// ---------------------------------------------------------------------------
// Host orchestration
// ---------------------------------------------------------------------------
struct Wts {
  const float *mass_w, *decay_w, *bv_q, *bv_k, *bv_v;
  const float *ln_mem_w, *ln_mem_b, *ln_attn_w, *ln_attn_b, *ctrl_ln_w, *ctrl_ln_b;
  const float *down_w, *down_b, *up_w, *up_b;
};

static void run_blocks(float* hb, float* weighted, float* tmp, float* inner,
                       float* attnbuf, float* logdec, float* cum,
                       _Float16* qkvb, _Float16* wtile, const Wts& W, hipStream_t stream) {
  for (int i = 0; i < LL; ++i) {
    gates_kernel<<<BT, 256, 0, stream>>>(hb, W.mass_w + (size_t)i * DD,
                                         W.decay_w + (size_t)i * DD, weighted, logdec);
    cumsum_kernel<<<BB, 1, 0, stream>>>(logdec, cum);
    mem_kernel<<<dim3(DD / 128, TT / 16, BB), 256, 0, stream>>>(weighted, cum, tmp);
    add_ln_kernel<<<BT, 256, 0, stream>>>(hb, tmp, W.ln_mem_w + (size_t)i * DD,
                                          W.ln_mem_b + (size_t)i * DD, nullptr, hb);
    ste_kernel<<<3 * HH, 256, 0, stream>>>(W.bv_q + (size_t)i * HH * HD,
                                           W.bv_k + (size_t)i * HH * HD,
                                           W.bv_v + (size_t)i * HH * HD, qkvb);
    attn_score_kernel<<<dim3(TT / 16, (TT / 16) / 8, BB * HH), 256, 0, stream>>>(hb, qkvb, attnbuf);
    attn_out_kernel<<<dim3(DD / 128, TT / 16, BB), 256, 0, stream>>>(hb, attnbuf, qkvb, tmp);
    add_ln_kernel<<<BT, 256, 0, stream>>>(hb, tmp, W.ln_attn_w + (size_t)i * DD,
                                          W.ln_attn_b + (size_t)i * DD, nullptr, hb);
    // Controller MLP
    add_ln_kernel<<<BT, 256, 0, stream>>>(hb, nullptr, W.ctrl_ln_w + (size_t)i * DD,
                                          W.ctrl_ln_b + (size_t)i * DD, nullptr, weighted);
    {
      size_t n4 = (size_t)DD * INNER / 4;
      convt_f16_kernel<<<(unsigned)((n4 + 255) / 256), 256, 0, stream>>>(
          W.down_w + (size_t)i * DD * INNER, wtile, DD, INNER);
    }
    gemm_kernel<<<dim3(INNER / 256, BT / 32), 256, 0, stream>>>(
        weighted, wtile, W.down_b + (size_t)i * INNER, inner, BT, INNER, DD, 1, nullptr);
    {
      size_t n4 = (size_t)INNER * DD / 4;
      convt_f16_kernel<<<(unsigned)((n4 + 255) / 256), 256, 0, stream>>>(
          W.up_w + (size_t)i * INNER * DD, wtile, INNER, DD);
    }
    gemm_kernel<<<dim3(DD / 256, BT / 32), 256, 0, stream>>>(
        inner, wtile, W.up_b + (size_t)i * DD, hb, BT, DD, INNER, 2, nullptr);
  }
}

extern "C" void kernel_launch(void* const* d_in, const int* in_sizes, int n_in,
                              void* d_out, int out_size, void* d_ws, size_t ws_size,
                              hipStream_t stream) {
  (void)in_sizes; (void)n_in; (void)out_size; (void)ws_size;
  const int*   idx        = (const int*)d_in[0];
  const float* codebook   = (const float*)d_in[1];
  Wts W;
  W.mass_w   = (const float*)d_in[2];
  W.decay_w  = (const float*)d_in[3];
  W.bv_q     = (const float*)d_in[4];
  W.bv_k     = (const float*)d_in[5];
  W.bv_v     = (const float*)d_in[6];
  W.ln_mem_w = (const float*)d_in[7];  W.ln_mem_b = (const float*)d_in[8];
  W.ln_attn_w= (const float*)d_in[9];  W.ln_attn_b= (const float*)d_in[10];
  W.ctrl_ln_w= (const float*)d_in[11]; W.ctrl_ln_b= (const float*)d_in[12];
  W.down_w   = (const float*)d_in[13]; W.down_b   = (const float*)d_in[14];
  W.up_w     = (const float*)d_in[15]; W.up_b     = (const float*)d_in[16];
  const float* thought_gates = (const float*)d_in[17];
  const float* thought_pos   = (const float*)d_in[18];
  const float* tl_ln_w  = (const float*)d_in[19];
  const float* tl_ln_b  = (const float*)d_in[20];
  const float* out_ln_w = (const float*)d_in[21];
  const float* out_ln_b = (const float*)d_in[22];
  const float* output_scale = (const float*)d_in[23];

  char* ws = (char*)d_ws;
  size_t off = 0;
  auto alloc = [&](size_t bytes) -> void* {
    void* p = ws + off;
    off += (bytes + 255) & ~(size_t)255;
    return p;
  };
  float* h0       = (float*)alloc((size_t)BT * DD * 4);
  float* h1       = (float*)alloc((size_t)BT * DD * 4);
  float* weighted = (float*)alloc((size_t)BT * DD * 4);
  float* tmp      = (float*)alloc((size_t)BT * DD * 4);
  float* inner    = (float*)alloc((size_t)BT * INNER * 4);
  float* attnbuf  = (float*)alloc((size_t)BB * HH * TT * TT * 4);
  float* logdec   = (float*)alloc((size_t)BB * TT * 4);
  float* cum      = (float*)alloc((size_t)BB * TT * 4);
  _Float16* qkvb  = (_Float16*)alloc((size_t)3 * HH * HD * 2);
  _Float16* wtile = (_Float16*)alloc((size_t)DD * INNER * 2);
  _Float16* cbf16 = (_Float16*)alloc((size_t)VV * DD * 2);

  // Embedding + first pass
  embed_kernel<<<BT, 256, 0, stream>>>(idx, codebook, h0);
  run_blocks(h0, weighted, tmp, inner, attnbuf, logdec, cum, qkvb, wtile, W, stream);

  // Thought pass (N_THOUGHTS=2 -> one extra pass with t=1)
  add_ln_kernel<<<BT, 256, 0, stream>>>(h0, nullptr, tl_ln_w, tl_ln_b,
                                        thought_pos + DD /* row 1 */, h1);
  run_blocks(h1, weighted, tmp, inner, attnbuf, logdec, cum, qkvb, wtile, W, stream);
  combine_kernel<<<(unsigned)(((size_t)BT * DD + 255) / 256), 256, 0, stream>>>(h0, h1, thought_gates);

  // Final LN + weight-tied logits GEMM
  add_ln_kernel<<<BT, 256, 0, stream>>>(h0, nullptr, out_ln_w, out_ln_b, nullptr, h1);
  {
    size_t n4 = (size_t)VV * DD / 4;
    conv_f16_kernel<<<(unsigned)((n4 + 255) / 256), 256, 0, stream>>>(codebook, cbf16, n4);
  }
  gemm_kernel<<<dim3(VV / 256, BT / 32), 256, 0, stream>>>(
      h1, cbf16, nullptr, (float*)d_out, BT, VV, DD, 0, output_scale);
}
